// NegUniform_49589692399688
// MI455X (gfx1250) — matrix operand order repaired and checked by reading, not compiled, not run on hardware
//
#include <hip/hip_runtime.h>
#include <hip/hip_bf16.h>
#include <math.h>

// ---------------------------------------------------------------------------
// NegUniform loss on MI455X (gfx1250, wave32, WMMA)
//   sims[j] = l2norm(feature) @ l2norm(negs[j])^T        (4x 4096x4096x128 GEMM)
//   mask same-class in set idx, per-row top-16, softmax-entropy -> scalar
// GEMM runs on v_wmma_f32_16x16x32_f16 (f16 in, f32 accumulate).
// ---------------------------------------------------------------------------

typedef __attribute__((ext_vector_type(16))) _Float16 v16h;
typedef __attribute__((ext_vector_type(8)))  float    v8f;

#define NN 4096
#define DD 128
#define JJ 4
#define KTOP 16
#define NEG_FILL_V (-1.0e9f)
#define NEG_INF_V  (-3.0e38f)

__device__ __forceinline__ float wave_sum32(float v) {
    for (int o = 16; o; o >>= 1) v += __shfl_xor(v, o, 32);
    return v;
}
__device__ __forceinline__ float wave_max32(float v) {
    for (int o = 16; o; o >>= 1) v = fmaxf(v, __shfl_xor(v, o, 32));
    return v;
}

// --------------------------------------------------------------------------
// Kernel 1: L2-normalize feature rows -> f16 row-major [N][D]
// One wave per row: 32 lanes x float4 = 128 elements.
// --------------------------------------------------------------------------
__global__ void norm_f_kernel(const float* __restrict__ feat,
                              _Float16* __restrict__ out) {
    int row  = (int)((blockIdx.x * blockDim.x + threadIdx.x) >> 5);
    int lane = (int)(threadIdx.x & 31);
    if (row >= NN) return;
    float4 v = reinterpret_cast<const float4*>(feat + (size_t)row * DD)[lane];
    float s = v.x * v.x + v.y * v.y + v.z * v.z + v.w * v.w;
    s = wave_sum32(s);
    float inv = 1.0f / fmaxf(sqrtf(s), 1e-12f);
    _Float16 h[4];
    h[0] = (_Float16)(v.x * inv);
    h[1] = (_Float16)(v.y * inv);
    h[2] = (_Float16)(v.z * inv);
    h[3] = (_Float16)(v.w * inv);
    reinterpret_cast<float2*>(out + (size_t)row * DD)[lane] =
        *reinterpret_cast<float2*>(h);
}

// --------------------------------------------------------------------------
// Kernel 2: L2-normalize negative rows and store TRANSPOSED f16 [J][D][N].
// Block = 256 threads (8 waves) handles one (j, 32-row m tile):
//   phase 1: each wave normalizes 4 rows into an LDS tile [m][d]
//   phase 2: coalesced transposed store: out[j][d][m]
// The [d][m] layout makes WMMA B-matrix loads coalesced: 16 lanes read 32B per k row.
// --------------------------------------------------------------------------
__global__ void norm_negs_T_kernel(const float* __restrict__ negs,
                                   _Float16* __restrict__ outT) {
    __shared__ _Float16 ld[32][130];  // pad to dodge bank conflicts
    int j    = (int)(blockIdx.x >> 7);          // 128 tiles of 32 rows per j
    int m0   = (int)((blockIdx.x & 127) << 5);  // *32
    int w    = (int)(threadIdx.x >> 5);
    int lane = (int)(threadIdx.x & 31);

    const float* base = negs + ((size_t)j * NN + m0) * DD;
    for (int rr = 0; rr < 4; ++rr) {
        int r = w * 4 + rr;
        float4 v = reinterpret_cast<const float4*>(base + (size_t)r * DD)[lane];
        float s = v.x * v.x + v.y * v.y + v.z * v.z + v.w * v.w;
        s = wave_sum32(s);
        float inv = 1.0f / fmaxf(sqrtf(s), 1e-12f);
        ld[r][4 * lane + 0] = (_Float16)(v.x * inv);
        ld[r][4 * lane + 1] = (_Float16)(v.y * inv);
        ld[r][4 * lane + 2] = (_Float16)(v.z * inv);
        ld[r][4 * lane + 3] = (_Float16)(v.w * inv);
    }
    __syncthreads();
    _Float16* dst = outT + (size_t)j * DD * NN + m0;
    for (int it = 0; it < 16; ++it) {
        int d = it * 8 + w;      // 0..127
        int m = lane;            // contiguous store across the wave
        dst[(size_t)d * NN + m] = ld[m][d];
    }
}

// --------------------------------------------------------------------------
// Kernel 3: WMMA similarity + per-row top-16.
// Grid = J * (N/16); block = 512 threads = 16 waves.
// Each block: one j, one 16-row tile of normalized features.
//   - A (16x128 f16) preloaded into 4 K-chunk register fragments per wave.
//     ISA layout: lane<16 row M=lane holds K {0..7,16..23}; lane>=16 holds
//     K {8..15,24..31} -> two b128 loads per chunk.
//   - Per tile: ALL four B fragments loaded first (64 strided f16 loads),
//     then the 4 chained v_wmma_f32_16x16x32_f16 drain with descending
//     s_wait_loadcnt, hiding most of the load latency.
//   - Same-class masking is branch-free: column target loaded
//     unconditionally, uniform (j==idx) folded into a -1 sentinel compare
//     value -> one v_cmp + v_cndmask per accumulator element.
//   - Loop over 8 column chunks of 512 candidates; per chunk each wave owns
//     one row and extracts top-16 (wave-max + ballot clear), banking
//     candidates; final top-16 over the 128 chunk winners -> [J][N][16].
// --------------------------------------------------------------------------
__global__ __launch_bounds__(512) void sims_topk_kernel(
    const _Float16* __restrict__ fh,     // [N][D] f16 normalized features
    const _Float16* __restrict__ negsT,  // [J][D][N] f16 normalized negatives
    const int* __restrict__ tgt,
    const int* __restrict__ idx_p,
    float* __restrict__ topout)          // [J][N][KTOP]
{
    __shared__ float sims[16][520];   // 16 rows x 512 cols (+pad)
    __shared__ float cand[16][128];   // 8 chunks x 16 candidates per row
    __shared__ int   tgtn[16];

    const int j    = (int)(blockIdx.x >> 8);          // 256 row-tiles per j
    const int n0   = (int)((blockIdx.x & 255) << 4);  // *16
    const int w    = (int)(threadIdx.x >> 5);
    const int lane = (int)(threadIdx.x & 31);
    const bool jeq = (j == idx_p[0]);

    if (threadIdx.x < 16) tgtn[threadIdx.x] = tgt[n0 + (int)threadIdx.x];

    // ---- preload A fragments (4 chunks of K=32) -------------------------
    union AV { v16h h; uint4 u[2]; };
    AV a[4];
    {
        const int row = n0 + (lane & 15);
        const int hi  = (lane >= 16);
        const _Float16* ap = fh + (size_t)row * DD;
        #pragma unroll
        for (int kc = 0; kc < 4; ++kc) {
            a[kc].u[0] = *reinterpret_cast<const uint4*>(ap + kc * 32 + (hi ? 8 : 0));
            a[kc].u[1] = *reinterpret_cast<const uint4*>(ap + kc * 32 + (hi ? 24 : 16));
        }
    }
    const _Float16* bbase = negsT + (size_t)j * DD * NN;
    __syncthreads();

    // row-target registers for this lane's 8 accumulator rows
    const int rbase = (lane >= 16) ? 8 : 0;
    int tn[8];
    #pragma unroll
    for (int r = 0; r < 8; ++r) tn[r] = tgtn[rbase + r];

    for (int c = 0; c < 8; ++c) {
        // ---- compute the 16x512 chunk ----------------------------------
        #pragma unroll
        for (int t2 = 0; t2 < 2; ++t2) {
            const int tl    = w * 2 + t2;            // tile in chunk: 0..31
            const int mbase = c * 512 + tl * 16;

            // issue all B loads first: one clause, then WMMAs drain it
            union BV { v16h h; uint4 u[2]; } b[4];
            #pragma unroll
            for (int kc = 0; kc < 4; ++kc) {
                const _Float16* bp = bbase + (size_t)(kc * 32) * NN + mbase + (lane & 15);
                const int bh = (lane >= 16) ? 8 : 0;
                #pragma unroll
                for (int s = 0; s < 16; ++s)
                    b[kc].h[s] = bp[(size_t)((s < 8) ? (bh + s) : (16 + bh + s - 8)) * NN];
            }
            // unconditional column-target load (no branch); sentinel select
            const int tm     = tgt[mbase + (lane & 15)];
            const int tm_eff = jeq ? tm : -1;        // uniform cndmask

            union CV { v8f v; float f[8]; } acc;
            acc.v = (v8f){0.f, 0.f, 0.f, 0.f, 0.f, 0.f, 0.f, 0.f};
            #pragma unroll
            for (int kc = 0; kc < 4; ++kc) {
                acc.v = __builtin_amdgcn_wmma_f32_16x16x32_f16(
                    false, a[kc].h, false, b[kc].h, (short)0, acc.v, false, false);
            }

            const int coll = tl * 16 + (lane & 15);
            #pragma unroll
            for (int r = 0; r < 8; ++r) {
                float vv = acc.f[r];
                vv = (tn[r] == tm_eff) ? NEG_FILL_V : vv;   // cmp + cndmask
                sims[rbase + r][coll] = vv;
            }
        }
        __syncthreads();

        // ---- chunk top-16: wave w owns row w ---------------------------
        {
            float vals[16];
            #pragma unroll
            for (int i = 0; i < 16; ++i) vals[i] = sims[w][lane + 32 * i];
            for (int t = 0; t < KTOP; ++t) {
                float lm = NEG_INF_V; int li = -1;
                #pragma unroll
                for (int i = 0; i < 16; ++i)
                    if (vals[i] > lm) { lm = vals[i]; li = i; }
                float wm = wave_max32(lm);
                unsigned long long bal = __ballot(lm == wm);
                int owner = __ffsll(bal) - 1;
                if (lane == owner && li >= 0) vals[li] = NEG_INF_V;
                if (lane == 0) cand[w][c * 16 + t] = wm;
            }
        }
        __syncthreads();
    }

    // ---- global top-16 from the 128 chunk candidates --------------------
    {
        float cv[4];
        #pragma unroll
        for (int i = 0; i < 4; ++i) cv[i] = cand[w][lane + 32 * i];
        float* outp = topout + ((size_t)j * NN + (n0 + w)) * KTOP;
        for (int t = 0; t < KTOP; ++t) {
            float lm = NEG_INF_V; int li = -1;
            #pragma unroll
            for (int i = 0; i < 4; ++i)
                if (cv[i] > lm) { lm = cv[i]; li = i; }
            float wm = wave_max32(lm);
            unsigned long long bal = __ballot(lm == wm);
            int owner = __ffsll(bal) - 1;
            if (lane == owner && li >= 0) cv[li] = NEG_INF_V;
            if (lane == 0) outp[t] = wm;
        }
    }
}

// --------------------------------------------------------------------------
// Kernel 4: per-row softmax entropy over J, geometric decay, mean + log(J).
// --------------------------------------------------------------------------
__global__ void loss_kernel(const float* __restrict__ top,  // [J][N][KTOP]
                            float* __restrict__ out) {
    __shared__ float red[256];
    const int n = (int)(blockIdx.x * blockDim.x + threadIdx.x);
    float acc = 0.0f;
    if (n < NN) {
        float decay[KTOP];
        float dsum = 0.0f, d = 1.0f;
        #pragma unroll
        for (int k = 0; k < KTOP; ++k) { decay[k] = d; dsum += d; d *= 0.95f; }
        float accrow = 0.0f;
        #pragma unroll
        for (int k = 0; k < KTOP; ++k) {
            float l[JJ];
            float mx = NEG_INF_V;
            #pragma unroll
            for (int jj = 0; jj < JJ; ++jj) {
                l[jj] = top[((size_t)jj * NN + n) * KTOP + k] * 100.0f;  // /TEMP
                mx = fmaxf(mx, l[jj]);
            }
            float s = 0.0f;
            #pragma unroll
            for (int jj = 0; jj < JJ; ++jj) s += expf(l[jj] - mx);
            float ls = logf(s);
            float ent = 0.0f;
            #pragma unroll
            for (int jj = 0; jj < JJ; ++jj) {
                float lp = l[jj] - mx - ls;
                ent += expf(lp) * lp;
            }
            accrow += ent * decay[k];
        }
        acc = accrow / dsum;
    }
    red[threadIdx.x] = acc;
    __syncthreads();
    for (int s = 128; s; s >>= 1) {
        if ((int)threadIdx.x < s) red[threadIdx.x] += red[threadIdx.x + s];
        __syncthreads();
    }
    if (threadIdx.x == 0) atomicAdd(out, red[0] / (float)NN);
    if (n == 0) atomicAdd(out, logf((float)JJ));
}

// --------------------------------------------------------------------------
extern "C" void kernel_launch(void* const* d_in, const int* in_sizes, int n_in,
                              void* d_out, int out_size, void* d_ws, size_t ws_size,
                              hipStream_t stream) {
    const float* feat  = (const float*)d_in[0];   // [4096,128] f32
    const int*   tgt   = (const int*)d_in[1];     // [4096] i32
    const float* negs  = (const float*)d_in[2];   // [4,4096,128] f32
    const int*   idx_p = (const int*)d_in[3];     // scalar
    float*       out   = (float*)d_out;

    char* ws = (char*)d_ws;
    _Float16* fh    = (_Float16*)(ws);                       // 1 MB
    _Float16* negsT = (_Float16*)(ws + (size_t)(1 << 20));   // 4 MB
    float*    top   = (float*)(ws + (size_t)(5 << 20));      // 1 MB

    hipMemsetAsync(d_out, 0, sizeof(float), stream);

    norm_f_kernel<<<NN / 8, 256, 0, stream>>>(feat, fh);
    norm_negs_T_kernel<<<JJ * (NN / 32), 256, 0, stream>>>(negs, negsT);
    sims_topk_kernel<<<JJ * (NN / 16), 512, 0, stream>>>(fh, negsT, tgt, idx_p, top);
    loss_kernel<<<NN / 256, 256, 0, stream>>>(top, out);
}
